// SpatialTemporalConv_58463094833483
// MI455X (gfx1250) — matrix operand order repaired and checked by reading, not compiled, not run on hardware
//
#include <hip/hip_runtime.h>
#include <math.h>
#include <stdint.h>

#define ID_LEN     207
#define INPUT_DIM  64
#define OUTPUT_DIM 128
#define WIN        3
#define BZ         16
#define SEQLEN     64
#define N_NODES    (BZ * ID_LEN)        // 3312
#define N_EDGES    (N_NODES * 16)       // 52992
#define CH         (ID_LEN * INPUT_DIM) // 13248
#define M_GEMM     (BZ * SEQLEN)        // 1024
#define N_GEMM     (OUTPUT_DIM * WIN)   // 384
#define K_GEMM     CH                   // 13248

// GEMM tiling
#define KSPLIT     3
#define NCHUNK     69                   // 13248 / 64 / 3
#define BK         64
#define RS         72                   // LDS row stride in elements (64 + TDM pad of 8)
#define ASZ        (64 * RS * 2)        // one 64-row tile buffer in bytes (9216)

typedef __attribute__((ext_vector_type(8)))  float   v8f;
typedef __attribute__((ext_vector_type(16))) __bf16  v16bf;
typedef __attribute__((ext_vector_type(8)))  __bf16  v8bf;
typedef unsigned int u32x4 __attribute__((ext_vector_type(4)));
typedef int          i32x4 __attribute__((ext_vector_type(4)));
typedef int          i32x8 __attribute__((ext_vector_type(8)));

// ---------------------------------------------------------------------------
// K1: per-node temporal mean, then q = (mean@Wq + bq)/sqrt(128), k = mean@Wk+bk
// ---------------------------------------------------------------------------
__global__ void k_mean_qk(const float* __restrict__ x,
                          const float* __restrict__ Wq, const float* __restrict__ bq,
                          const float* __restrict__ Wk, const float* __restrict__ bk,
                          float* __restrict__ q, float* __restrict__ k) {
    __shared__ float mean[INPUT_DIM];
    const int n = blockIdx.x;
    const int b = n / ID_LEN, id = n % ID_LEN;
    const int tid = threadIdx.x;
    if (tid < INPUT_DIM) {
        const float* p = x + (size_t)b * SEQLEN * CH + (size_t)id * INPUT_DIM + tid;
        float s = 0.f;
        #pragma unroll 4
        for (int t = 0; t < SEQLEN; ++t) s += p[(size_t)t * CH];
        mean[tid] = s * (1.0f / SEQLEN);
    }
    __syncthreads();
    const int o = tid;  // 0..127
    float aq = 0.f, ak = 0.f;
    #pragma unroll 4
    for (int d = 0; d < INPUT_DIM; ++d) {
        const float m = mean[d];
        aq = fmaf(m, Wq[d * OUTPUT_DIM + o], aq);
        ak = fmaf(m, Wk[d * OUTPUT_DIM + o], ak);
    }
    q[(size_t)n * OUTPUT_DIM + o] = (aq + bq[o]) * 0.08838834764831845f; // 1/sqrt(128)
    k[(size_t)n * OUTPUT_DIM + o] = ak + bk[o];
}

// ---------------------------------------------------------------------------
// K2: per-edge score = dot(k[src], q[dst])   (q already scaled)
// ---------------------------------------------------------------------------
__global__ void k_scores(const float* __restrict__ q, const float* __restrict__ k,
                         const int* __restrict__ esrc, const int* __restrict__ edst,
                         float* __restrict__ scores) {
    const int e = blockIdx.x * blockDim.x + threadIdx.x;
    if (e >= N_EDGES) return;
    const float* kr = k + (size_t)esrc[e] * OUTPUT_DIM;
    const float* qr = q + (size_t)edst[e] * OUTPUT_DIM;
    float s = 0.f;
    #pragma unroll 8
    for (int o = 0; o < OUTPUT_DIM; ++o) s = fmaf(kr[o], qr[o], s);
    scores[e] = s;
}

// ---------------------------------------------------------------------------
// CSR build (deterministic per-dst processing, no float atomics)
// ---------------------------------------------------------------------------
__global__ void k_zero_i32(int* __restrict__ p, int n) {
    int i = blockIdx.x * blockDim.x + threadIdx.x;
    if (i < n) p[i] = 0;
}
__global__ void k_count(const int* __restrict__ edst, int* __restrict__ deg,
                        int* __restrict__ slot) {
    int e = blockIdx.x * blockDim.x + threadIdx.x;
    if (e >= N_EDGES) return;
    slot[e] = atomicAdd(&deg[edst[e]], 1);
}
__global__ void k_scan(const int* __restrict__ deg, int* __restrict__ offsets) {
    __shared__ int part[256];
    const int tid = threadIdx.x;
    const int CHUNK = (N_NODES + 255) / 256; // 13
    int s = 0;
    for (int j = 0; j < CHUNK; ++j) {
        int idx = tid * CHUNK + j;
        if (idx < N_NODES) s += deg[idx];
    }
    part[tid] = s;
    __syncthreads();
    if (tid == 0) {
        int run = 0;
        for (int i = 0; i < 256; ++i) { int t = part[i]; part[i] = run; run += t; }
        offsets[N_NODES] = run;
    }
    __syncthreads();
    int run = part[tid];
    for (int j = 0; j < CHUNK; ++j) {
        int idx = tid * CHUNK + j;
        if (idx < N_NODES) { offsets[idx] = run; run += deg[idx]; }
    }
}
__global__ void k_scatter(const int* __restrict__ edst, const int* __restrict__ offsets,
                          const int* __restrict__ slot, int* __restrict__ perm) {
    int e = blockIdx.x * blockDim.x + threadIdx.x;
    if (e >= N_EDGES) return;
    perm[offsets[edst[e]] + slot[e]] = e;
}

// ---------------------------------------------------------------------------
// K4: per-dst segment softmax + weighted gather of v[src] (= rows of x),
// writes GEMM A-matrix directly in bf16:  A[(b*64+s)][id*64+d]
// ---------------------------------------------------------------------------
__global__ void k_aggregate(const float* __restrict__ x,
                            const float* __restrict__ scores,
                            const int* __restrict__ esrc,
                            const int* __restrict__ offsets,
                            const int* __restrict__ perm,
                            __bf16* __restrict__ A) {
    __shared__ float red[256];
    __shared__ float s_mx, s_den;
    const int n = blockIdx.x;
    const int b = n / ID_LEN, id = n % ID_LEN;
    const int tid = threadIdx.x;
    const int start = offsets[n], end = offsets[n + 1];
    // --- max ---
    float mx = -INFINITY;
    for (int i = start + tid; i < end; i += 256) mx = fmaxf(mx, scores[perm[i]]);
    red[tid] = mx; __syncthreads();
    for (int w = 128; w > 0; w >>= 1) {
        if (tid < w) red[tid] = fmaxf(red[tid], red[tid + w]);
        __syncthreads();
    }
    if (tid == 0) s_mx = red[0];
    __syncthreads();
    const float bmx = s_mx;
    // --- denom ---
    float dn = 0.f;
    for (int i = start + tid; i < end; i += 256) dn += __expf(scores[perm[i]] - bmx);
    red[tid] = dn; __syncthreads();
    for (int w = 128; w > 0; w >>= 1) {
        if (tid < w) red[tid] += red[tid + w];
        __syncthreads();
    }
    if (tid == 0) s_den = red[0];
    __syncthreads();
    const float inv = (end > start) ? 1.0f / s_den : 0.0f;
    // --- weighted gather: thread owns cols {tid + 256*j}, col = s*64+d ---
    const int d  = tid & 63;
    const int s0 = tid >> 6;
    float acc[16];
    #pragma unroll
    for (int j = 0; j < 16; ++j) acc[j] = 0.f;
    for (int i = start; i < end; ++i) {
        const int e = perm[i];
        const float a = __expf(scores[e] - bmx) * inv;
        const int src = esrc[e];
        const int sb = src / ID_LEN, sid = src % ID_LEN;
        const float* vb = x + (size_t)sb * SEQLEN * CH + (size_t)sid * INPUT_DIM + d;
        #pragma unroll
        for (int j = 0; j < 16; ++j)
            acc[j] = fmaf(a, vb[(size_t)(s0 + 4 * j) * CH], acc[j]);
    }
    #pragma unroll
    for (int j = 0; j < 16; ++j) {
        const int s = s0 + 4 * j;
        A[(size_t)(b * SEQLEN + s) * K_GEMM + (size_t)id * INPUT_DIM + d] = (__bf16)acc[j];
    }
}

// ---------------------------------------------------------------------------
// K5: conv weights -> bf16 "column-major" B:  Bt[(o*3+w)][c] = conv_w[o][c][w]
// ---------------------------------------------------------------------------
__global__ void k_wconv(const float* __restrict__ conv_w, __bf16* __restrict__ Bt) {
    const int gid = blockIdx.x * blockDim.x + threadIdx.x;
    if (gid >= N_GEMM * K_GEMM) return;
    const int nidx = gid / K_GEMM;       // o*3+w
    const int c    = gid % K_GEMM;
    const int o = nidx / WIN, w = nidx % WIN;
    Bt[gid] = (__bf16)conv_w[((size_t)o * K_GEMM + c) * WIN + w];
}

// ---------------------------------------------------------------------------
// TDM: issue a 2D 64x64 (bf16) tile load into LDS, with row padding of
// 4 DWORDs per 32 DWORDs (-> 144 B LDS row stride, bank-conflict-free frags).
// Descriptor per CDNA5 ISA ch.8: group0 = {count|lds_addr|global_addr|type=2},
// group1 = {data_size=2B, pad_enable, pad_interval=code4(32dw),
//           pad_amount=code3(4dw), tensor dims/strides, tile 64x64}.
// ---------------------------------------------------------------------------
__device__ __forceinline__ void tdm_load_tile_2d(const __bf16* gbase,
                                                 unsigned lds_off,
                                                 unsigned tensor_rows) {
    unsigned long long ga = (unsigned long long)(uintptr_t)gbase;
    u32x4 g0;
    g0[0] = 1u;                                   // count=1, user descriptor
    g0[1] = lds_off;                              // LDS byte address
    g0[2] = (unsigned)ga;                         // global_addr[31:0]
    g0[3] = (unsigned)(ga >> 32) | (2u << 30);    // global_addr[56:32] | type=2
    i32x8 g1;
    g1[0] = (int)((1u << 16) | (1u << 20) | (4u << 22) | (3u << 25));
    g1[1] = (int)(((unsigned)K_GEMM & 0xFFFFu) << 16);            // tensor_dim0 lo16
    g1[2] = (int)(((unsigned)K_GEMM >> 16) | ((tensor_rows & 0xFFFFu) << 16));
    g1[3] = (int)((tensor_rows >> 16) | (64u << 16));             // tile_dim0 = 64
    g1[4] = 64;                                                   // tile_dim1=64, tile_dim2=0
    g1[5] = (int)K_GEMM;                                          // dim0_stride lo32
    g1[6] = (int)(((unsigned)K_GEMM & 0xFFFFu) << 16);            // dim0_str hi=0 | dim1_str lo16
    g1[7] = 0;
    i32x4 gz = {0, 0, 0, 0};
#if __clang_major__ >= 23
    i32x8 gz8 = {0, 0, 0, 0, 0, 0, 0, 0};
    __builtin_amdgcn_tensor_load_to_lds(g0, g1, gz, gz, gz8, 0);
#else
    __builtin_amdgcn_tensor_load_to_lds(g0, g1, gz, gz, 0);
#endif
}

// ---------------------------------------------------------------------------
// K6: WMMA GEMM  Z[m][nidx] = sum_c A[m][c] * Bt[nidx][c]
// 4 waves / block, 64x64 tile, K-split over gridDim.z.
// Double-buffered 64x64 A/B tiles staged in LDS by the Tensor Data Mover
// (wave0 -> A, wave1 -> B; producer waits TENSORcnt, barrier publishes).
// Fragment striping (contiguous K, same order in A and B):
//   A (16x32): lane%16 = row, 16 contiguous K at base (lane/16)*16
//   B (32x16): lane%16 = column, 16 contiguous K at base (lane/16)*16
// ---------------------------------------------------------------------------
__global__ void __launch_bounds__(128)
k_wmma_gemm(const __bf16* __restrict__ A, const __bf16* __restrict__ Bt,
            float* __restrict__ Z) {
    extern __shared__ char smem[];                   // 4 * ASZ bytes, base offset 0
    const int wv   = threadIdx.x >> 5;
    const int lane = threadIdx.x & 31;
    const int m0b  = blockIdx.x * 64;
    const int n0b  = blockIdx.y * 64;
    const int kbase = blockIdx.z * NCHUNK * BK;

    const __bf16* aTile = A  + (size_t)m0b * K_GEMM + kbase;
    const __bf16* bTile = Bt + (size_t)n0b * K_GEMM + kbase;

    const int wm   = (wv >> 1) * 32;                 // wave's M offset in tile
    const int wn   = (wv & 1) * 32;                  // wave's N offset in tile
    const int l15  = lane & 15;
    const int off0 = (lane < 16) ? 0 : 16;           // A K-offset pattern
    const int bko  = (lane >> 4) << 4;               // B K-base pattern

    v8f acc00 = {}, acc01 = {}, acc10 = {}, acc11 = {};

    // prologue: stage chunk 0
    if (wv == 0)      tdm_load_tile_2d(aTile, 0u, (unsigned)M_GEMM);
    else if (wv == 1) tdm_load_tile_2d(bTile, 2u * ASZ, (unsigned)N_GEMM);

    for (int c = 0; c < NCHUNK; ++c) {
        const int cur = c & 1;
        if (wv == 0) {
            if (c + 1 < NCHUNK) {
                tdm_load_tile_2d(aTile + (size_t)(c + 1) * BK, (unsigned)((1 - cur) * ASZ),
                                 (unsigned)M_GEMM);
                __builtin_amdgcn_s_wait_tensorcnt(1);
            } else {
                __builtin_amdgcn_s_wait_tensorcnt(0);
            }
        } else if (wv == 1) {
            if (c + 1 < NCHUNK) {
                tdm_load_tile_2d(bTile + (size_t)(c + 1) * BK,
                                 (unsigned)(2 * ASZ + (1 - cur) * ASZ), (unsigned)N_GEMM);
                __builtin_amdgcn_s_wait_tensorcnt(1);
            } else {
                __builtin_amdgcn_s_wait_tensorcnt(0);
            }
        }
        __syncthreads();                             // buffer `cur` ready for everyone

        const __bf16* Ab = (const __bf16*)(smem + cur * ASZ);
        const __bf16* Bb = (const __bf16*)(smem + 2 * ASZ + cur * ASZ);
        const int ra = wm + l15;
        const int rb = wn + l15;
        #pragma unroll
        for (int kc = 0; kc < BK; kc += 32) {
            v8bf alo0 = *(const v8bf*)(Ab + ra * RS + kc + off0);
            v8bf ahi0 = *(const v8bf*)(Ab + ra * RS + kc + off0 + 8);
            v8bf alo1 = *(const v8bf*)(Ab + (ra + 16) * RS + kc + off0);
            v8bf ahi1 = *(const v8bf*)(Ab + (ra + 16) * RS + kc + off0 + 8);
            v16bf a0 = __builtin_shufflevector(alo0, ahi0, 0,1,2,3,4,5,6,7,8,9,10,11,12,13,14,15);
            v16bf a1 = __builtin_shufflevector(alo1, ahi1, 0,1,2,3,4,5,6,7,8,9,10,11,12,13,14,15);
            v8bf blo0 = *(const v8bf*)(Bb + rb * RS + kc + bko);
            v8bf bhi0 = *(const v8bf*)(Bb + rb * RS + kc + bko + 8);
            v8bf blo1 = *(const v8bf*)(Bb + (rb + 16) * RS + kc + bko);
            v8bf bhi1 = *(const v8bf*)(Bb + (rb + 16) * RS + kc + bko + 8);
            v16bf b0 = __builtin_shufflevector(blo0, bhi0, 0,1,2,3,4,5,6,7,8,9,10,11,12,13,14,15);
            v16bf b1 = __builtin_shufflevector(blo1, bhi1, 0,1,2,3,4,5,6,7,8,9,10,11,12,13,14,15);
            acc00 = __builtin_amdgcn_wmma_f32_16x16x32_bf16(false, a0, false, b0, (short)0, acc00, false, false);
            acc01 = __builtin_amdgcn_wmma_f32_16x16x32_bf16(false, a0, false, b1, (short)0, acc01, false, false);
            acc10 = __builtin_amdgcn_wmma_f32_16x16x32_bf16(false, a1, false, b0, (short)0, acc10, false, false);
            acc11 = __builtin_amdgcn_wmma_f32_16x16x32_bf16(false, a1, false, b1, (short)0, acc11, false, false);
        }
        __syncthreads();                             // done reading `cur`; producer may overwrite
    }

    // C/D layout: VGPR i -> M = i + (lane&16 ? 8 : 0), N = lane&15
    float* Zp = Z + (size_t)blockIdx.z * M_GEMM * N_GEMM;
    const int lrow = (lane & 16) ? 8 : 0;
    const int m0 = m0b + wm, n0 = n0b + wn;
    #pragma unroll
    for (int i = 0; i < 8; ++i) {
        const int r0 = m0 + i + lrow, r1 = r0 + 16;
        Zp[(size_t)r0 * N_GEMM + n0 + l15]      = acc00[i];
        Zp[(size_t)r0 * N_GEMM + n0 + 16 + l15] = acc01[i];
        Zp[(size_t)r1 * N_GEMM + n0 + l15]      = acc10[i];
        Zp[(size_t)r1 * N_GEMM + n0 + 16 + l15] = acc11[i];
    }
}

// ---------------------------------------------------------------------------
// K7: reduce K-split partials + 3-tap shifted combine
// ---------------------------------------------------------------------------
__global__ void k_epilogue(const float* __restrict__ Z, float* __restrict__ out) {
    const int idx = blockIdx.x * blockDim.x + threadIdx.x;
    if (idx >= BZ * SEQLEN * OUTPUT_DIM) return;
    const int b = idx / (SEQLEN * OUTPUT_DIM);
    const int r = idx % (SEQLEN * OUTPUT_DIM);
    const int t = r / OUTPUT_DIM, o = r % OUTPUT_DIM;
    float acc = 0.f;
    #pragma unroll
    for (int z = 0; z < KSPLIT; ++z) {
        const float* Zp = Z + (size_t)z * M_GEMM * N_GEMM;
        acc += Zp[(size_t)(b * SEQLEN + t) * N_GEMM + o * WIN + 1];
        if (t > 0)          acc += Zp[(size_t)(b * SEQLEN + t - 1) * N_GEMM + o * WIN + 0];
        if (t < SEQLEN - 1) acc += Zp[(size_t)(b * SEQLEN + t + 1) * N_GEMM + o * WIN + 2];
    }
    out[idx] = acc;
}

// ---------------------------------------------------------------------------
static inline size_t rnd256(size_t x) { return (x + 255) & ~(size_t)255; }

extern "C" void kernel_launch(void* const* d_in, const int* in_sizes, int n_in,
                              void* d_out, int out_size, void* d_ws, size_t ws_size,
                              hipStream_t stream) {
    const float* x      = (const float*)d_in[0];
    const float* Wq     = (const float*)d_in[1];
    const float* bq     = (const float*)d_in[2];
    const float* Wk     = (const float*)d_in[3];
    const float* bk     = (const float*)d_in[4];
    const float* conv_w = (const float*)d_in[5];
    const int*   esrc   = (const int*)d_in[6];
    const int*   edst   = (const int*)d_in[7];
    float* out = (float*)d_out;

    // workspace carve (~48 MB total)
    char* ws = (char*)d_ws;
    float* q       = (float*)ws;  ws += rnd256((size_t)N_NODES * OUTPUT_DIM * 4);
    float* k       = (float*)ws;  ws += rnd256((size_t)N_NODES * OUTPUT_DIM * 4);
    float* scores  = (float*)ws;  ws += rnd256((size_t)N_EDGES * 4);
    int*   deg     = (int*)ws;    ws += rnd256((size_t)N_NODES * 4);
    int*   offsets = (int*)ws;    ws += rnd256((size_t)(N_NODES + 1) * 4);
    int*   slot    = (int*)ws;    ws += rnd256((size_t)N_EDGES * 4);
    int*   perm    = (int*)ws;    ws += rnd256((size_t)N_EDGES * 4);
    __bf16* Abf    = (__bf16*)ws; ws += rnd256((size_t)M_GEMM * K_GEMM * 2);
    __bf16* Btbf   = (__bf16*)ws; ws += rnd256((size_t)N_GEMM * K_GEMM * 2);
    float* Z       = (float*)ws;  ws += rnd256((size_t)KSPLIT * M_GEMM * N_GEMM * 4);

    // 1) mean + q/k projections
    k_mean_qk<<<N_NODES, 128, 0, stream>>>(x, Wq, bq, Wk, bk, q, k);
    // 2) edge scores
    k_scores<<<N_EDGES / 256, 256, 0, stream>>>(q, k, esrc, edst, scores);
    // 3) CSR build
    k_zero_i32<<<(N_NODES + 255) / 256, 256, 0, stream>>>(deg, N_NODES);
    k_count<<<N_EDGES / 256, 256, 0, stream>>>(edst, deg, slot);
    k_scan<<<1, 256, 0, stream>>>(deg, offsets);
    k_scatter<<<N_EDGES / 256, 256, 0, stream>>>(edst, offsets, slot, perm);
    // 4) segment softmax + weighted gather -> bf16 A
    k_aggregate<<<N_NODES, 256, 0, stream>>>(x, scores, esrc, offsets, perm, Abf);
    // 5) conv weights -> bf16 B (column-major by (o,w))
    k_wconv<<<(N_GEMM * K_GEMM) / 256, 256, 0, stream>>>(conv_w, Btbf);
    // 6) WMMA GEMM with TDM-staged LDS tiles: 1024 x 13248 x 384, K-split 3
    dim3 gg(M_GEMM / 64, N_GEMM / 64, KSPLIT);
    k_wmma_gemm<<<gg, 128, 4 * ASZ, stream>>>(Abf, Btbf, Z);
    // 7) K-split reduce + 3-tap epilogue
    k_epilogue<<<(BZ * SEQLEN * OUTPUT_DIM) / 256, 256, 0, stream>>>(Z, out);
}
